// UniformQuantAttention_54631984005704
// MI455X (gfx1250) — matrix operand order repaired and checked by reading, not compiled, not run on hardware
//
#include <hip/hip_runtime.h>

typedef unsigned short u16;
typedef __attribute__((ext_vector_type(16))) __bf16 v16bf;
typedef __attribute__((ext_vector_type(8)))  float  v8f;
typedef __attribute__((ext_vector_type(4)))  unsigned int v4u;
typedef __attribute__((ext_vector_type(8)))  int v8i;
typedef __attribute__((ext_vector_type(4)))  int v4i;

constexpr int SEQ = 2048;
constexpr int EMB = 2048;
constexpr int NH  = 16;
constexpr int HD  = 128;                       // head dim
constexpr float SCALE = 0.0883883476483184f;   // 1/sqrt(128)

// ---------------------------------------------------------------- TDM support
#if defined(__has_builtin)
# if __has_builtin(__builtin_amdgcn_tensor_load_to_lds)
#  define HAVE_TDM 1
# endif
#endif
#ifndef HAVE_TDM
# define HAVE_TDM 0
#endif

#if HAVE_TDM
__device__ __forceinline__ void tdm_issue(v4u g0, v8i g1) {
  v4i z4 = {0, 0, 0, 0};
#if __clang_major__ >= 23
  v8i z8 = {0, 0, 0, 0, 0, 0, 0, 0};
  __builtin_amdgcn_tensor_load_to_lds(g0, g1, z4, z4, z8, 0);
#else
  __builtin_amdgcn_tensor_load_to_lds(g0, g1, z4, z4, 0);
#endif
}

// 2-D bf16 tile DMA: `rows` rows of `cols` contiguous elements, source row
// stride `stride` elements, packed contiguously at LDS byte offset lds_off.
__device__ __forceinline__ void tdm_load_tile(unsigned lds_off, const void* gptr,
                                              int cols, int rows, int stride) {
  unsigned long long ga = (unsigned long long)(uintptr_t)gptr;
  v4u g0 = { 1u,                                   // count=1 (valid user D#)
             lds_off,                               // LDS byte address
             (unsigned)ga,                          // global_addr[31:0]
             ((unsigned)(ga >> 32) & 0x1FFFFFFu) | (2u << 30) }; // [56:32]|type=2
  v8i g1 = { (int)(1u << 16),                      // data_size=1 -> 2 bytes
             (int)((unsigned)cols << 16),          // tensor_dim0[15:0]
             (int)((unsigned)rows << 16),          // tensor_dim1[15:0]
             (int)((unsigned)cols << 16),          // tile_dim0
             (int)rows,                            // tile_dim1 (tile_dim2=0)
             stride,                               // tensor_dim0_stride[31:0]
             0, 0 };
  tdm_issue(g0, g1);
}

__device__ __forceinline__ void tdm_wait(int n) {
  if (n == 0)      __builtin_amdgcn_s_wait_tensorcnt(0);
  else if (n == 1) __builtin_amdgcn_s_wait_tensorcnt(1);
  else             __builtin_amdgcn_s_wait_tensorcnt(2);
}
#endif  // HAVE_TDM

// ------------------------------------------------------------------- helpers
__device__ __forceinline__ u16 f2bf(float x) {
  unsigned u = __float_as_uint(x);
  u += 0x7FFFu + ((u >> 16) & 1u);             // round-to-nearest-even
  return (u16)(u >> 16);
}

union AFrag { v16bf v; uint4 q[2]; };

// 16-bit A-matrix fragment (16x32, MxK): lane holds row (lane&15);
// element i = K kb+i, kb=(lane&16)?16:0.
__device__ __forceinline__ v16bf load_a_frag(const u16* rowp, int lane) {
  const int kb = (lane & 16) ? 16 : 0;
  AFrag f;
  f.q[0] = *(const uint4*)(rowp + kb);
  f.q[1] = *(const uint4*)(rowp + kb + 8);
  return f.v;
}

// 16-bit B-matrix fragment (32x16, KxN): lane holds column (lane&15);
// element i = K ((lane&16)+i). Caller passes a contiguous-K pointer already
// offset by (lane&16).
__device__ __forceinline__ v16bf load_b_frag(const u16* p_colK) {
  AFrag f;
  f.q[0] = *(const uint4*)(p_colK);
  f.q[1] = *(const uint4*)(p_colK + 8);
  return f.v;
}

__device__ __forceinline__ v8f wmma_bf16(v16bf a, v16bf b, v8f c) {
  return __builtin_amdgcn_wmma_f32_16x16x32_bf16(
      false, a, false, b, (short)0, c, false, false);
}

// DPP16 butterfly reduction within each 16-lane row (pure VALU, no LDS).
template <int CTRL>
__device__ __forceinline__ float dppmov(float x) {
  return __int_as_float(__builtin_amdgcn_update_dpp(
      0, __float_as_int(x), CTRL, 0xF, 0xF, true));
}
__device__ __forceinline__ float red16_max(float x) {
  x = fmaxf(x, dppmov<0xB1>(x));   // quad_perm [1,0,3,2]  (xor 1)
  x = fmaxf(x, dppmov<0x4E>(x));   // quad_perm [2,3,0,1]  (xor 2)
  x = fmaxf(x, dppmov<0x141>(x));  // row_half_mirror      (xor 4 once quads uniform)
  x = fmaxf(x, dppmov<0x140>(x));  // row_mirror           (xor 8 once octets uniform)
  return x;
}
__device__ __forceinline__ float red16_sum(float x) {
  x += dppmov<0xB1>(x);
  x += dppmov<0x4E>(x);
  x += dppmov<0x141>(x);
  x += dppmov<0x140>(x);
  return x;
}

// ---------------------------------------------------------------- fp32 -> bf16
__global__ void __launch_bounds__(256)
cvt_f32_bf16(const float* __restrict__ in, u16* __restrict__ out, int n) {
  int i = (blockIdx.x * 256 + threadIdx.x) * 4;
  if (i + 3 < n) {
    float4 v = *(const float4*)(in + i);
    uint2 p;
    p.x = (unsigned)f2bf(v.x) | ((unsigned)f2bf(v.y) << 16);
    p.y = (unsigned)f2bf(v.z) | ((unsigned)f2bf(v.w) << 16);
    *(uint2*)(out + i) = p;
  } else {
    for (; i < n; ++i) out[i] = f2bf(in[i]);
  }
}

// --------------------------------------------- C[M,N] = A[M,K] * W[N,K]^T + b
// Block: 256 threads = 8 waves (4 along M x 2 along N), block tile 128x128,
// each wave a 32x64 register tile (2x4 WMMA accumulators -> 8 WMMAs per
// K-step from 12 ds_load_b128, 1.5 DS/WMMA).
// TDM path: A (128x32) and W (128x32) K-step tiles double-buffered in LDS,
// DMA'd one K-step ahead by the Tensor Data Mover while WMMAs run.
template <bool OUT_BF16>
__global__ void __launch_bounds__(256)
gemm_xt_wt(const u16* __restrict__ A, const u16* __restrict__ W,
           const float* __restrict__ bias, void* __restrict__ Cout,
           int M, int N, int K) {
  const int lane = threadIdx.x & 31;
  const int w    = threadIdx.x >> 5;
  const int wm   = w & 3;                  // 4 waves along M
  const int wn   = w >> 2;                 // 2 waves along N
  const int row0 = blockIdx.y * 128 + wm * 32;
  const int col0 = blockIdx.x * 128 + wn * 64;
  const int ar   = lane & 15;
  const int bsel = lane & 16;

  v8f acc[2][4] = {};

#if HAVE_TDM
  __shared__ alignas(16) u16 As[2][128][32];   // 16 KB
  __shared__ alignas(16) u16 Ws[2][128][32];   // 16 KB
  const u16* Ablk = A + (size_t)(blockIdx.y * 128) * K;
  const u16* Wblk = W + (size_t)(blockIdx.x * 128) * K;
  if (w == 0) {
    tdm_load_tile((unsigned)(uintptr_t)(void*)&As[0][0][0], Ablk, 32, 128, K);
    tdm_load_tile((unsigned)(uintptr_t)(void*)&Ws[0][0][0], Wblk, 32, 128, K);
  }
  const int asub = wm * 32;
  const int wsub = wn * 64;
  for (int it = 0; it < K / 32; ++it) {
    const int k0  = it * 32;
    const int cur = it & 1;
    __syncthreads();                        // prev iteration done with buf cur^1
    if (w == 0) {
      if (k0 + 32 < K) {
        tdm_load_tile((unsigned)(uintptr_t)(void*)&As[cur ^ 1][0][0],
                      Ablk + (k0 + 32), 32, 128, K);
        tdm_load_tile((unsigned)(uintptr_t)(void*)&Ws[cur ^ 1][0][0],
                      Wblk + (k0 + 32), 32, 128, K);
        tdm_wait(2);                        // current pair landed, next streaming
      } else {
        tdm_wait(0);
      }
    }
    __syncthreads();
    v16bf a0 = load_a_frag(&As[cur][asub +  0 + ar][0], lane);
    v16bf a1 = load_a_frag(&As[cur][asub + 16 + ar][0], lane);
    v16bf b0 = load_b_frag(&Ws[cur][wsub +  0 + ar][bsel]);
    v16bf b1 = load_b_frag(&Ws[cur][wsub + 16 + ar][bsel]);
    v16bf b2 = load_b_frag(&Ws[cur][wsub + 32 + ar][bsel]);
    v16bf b3 = load_b_frag(&Ws[cur][wsub + 48 + ar][bsel]);
    acc[0][0] = wmma_bf16(a0, b0, acc[0][0]);
    acc[0][1] = wmma_bf16(a0, b1, acc[0][1]);
    acc[0][2] = wmma_bf16(a0, b2, acc[0][2]);
    acc[0][3] = wmma_bf16(a0, b3, acc[0][3]);
    acc[1][0] = wmma_bf16(a1, b0, acc[1][0]);
    acc[1][1] = wmma_bf16(a1, b1, acc[1][1]);
    acc[1][2] = wmma_bf16(a1, b2, acc[1][2]);
    acc[1][3] = wmma_bf16(a1, b3, acc[1][3]);
  }
#else
  for (int k0 = 0; k0 < K; k0 += 32) {
    v16bf a0 = load_a_frag(A + (size_t)(row0 +  0 + ar) * K + k0, lane);
    v16bf a1 = load_a_frag(A + (size_t)(row0 + 16 + ar) * K + k0, lane);
    v16bf b0 = load_b_frag(W + (size_t)(col0 +  0 + ar) * K + k0 + bsel);
    v16bf b1 = load_b_frag(W + (size_t)(col0 + 16 + ar) * K + k0 + bsel);
    v16bf b2 = load_b_frag(W + (size_t)(col0 + 32 + ar) * K + k0 + bsel);
    v16bf b3 = load_b_frag(W + (size_t)(col0 + 48 + ar) * K + k0 + bsel);
    acc[0][0] = wmma_bf16(a0, b0, acc[0][0]);
    acc[0][1] = wmma_bf16(a0, b1, acc[0][1]);
    acc[0][2] = wmma_bf16(a0, b2, acc[0][2]);
    acc[0][3] = wmma_bf16(a0, b3, acc[0][3]);
    acc[1][0] = wmma_bf16(a1, b0, acc[1][0]);
    acc[1][1] = wmma_bf16(a1, b1, acc[1][1]);
    acc[1][2] = wmma_bf16(a1, b2, acc[1][2]);
    acc[1][3] = wmma_bf16(a1, b3, acc[1][3]);
  }
#endif

  // C layout: VGPR j -> row j (lanes 0-15) / row j+8 (lanes 16-31), col lane&15.
  const int half8 = (lane >> 4) << 3;
  const int c = lane & 15;
  #pragma unroll
  for (int mt = 0; mt < 2; ++mt)
    #pragma unroll
    for (int nt = 0; nt < 4; ++nt)
      #pragma unroll
      for (int j = 0; j < 8; ++j) {
        int r  = row0 + mt * 16 + j + half8;
        int cc = col0 + nt * 16 + c;
        float val = acc[mt][nt][j] + bias[cc];
        if (OUT_BF16) ((u16*)Cout)[(size_t)r * N + cc]  = f2bf(val);
        else          ((float*)Cout)[(size_t)r * N + cc] = val;
      }
}

// ------------------------------------------------------- flash attention (bf16)
// Grid: (SEQ/128, NH). Block: 256 threads = 8 waves, each wave owns 16 query
// rows of one head. K tiles (32x128) TDM-DMA'd one iteration ahead into a
// double buffer; V staged transposed [d][k] so PV B-fragments are contiguous.
__global__ void __launch_bounds__(256)
attn_flash(const u16* __restrict__ Qb, const u16* __restrict__ Kb,
           const u16* __restrict__ Vb, const float* __restrict__ mask,
           u16* __restrict__ Ctx) {
  __shared__ alignas(16) u16 Kt[2][32][HD];    // [buf][key][d]  16 KB
  __shared__ alignas(16) u16 Vt[HD][32];       // [d][key]        8 KB
  __shared__ alignas(16) u16 Pp[8][16 * 32];   // per-wave P tile 8 KB

  const int tid   = threadIdx.x;
  const int lane  = tid & 31;
  const int w     = tid >> 5;
  const int head  = blockIdx.y;
  const int q0    = blockIdx.x * 128 + w * 16;
  const int hoff  = head * HD;
  const int c     = lane & 15;
  const int half8 = (lane >> 4) << 3;
  const int bsel  = lane & 16;

  // Q fragments for this wave's 16 rows (K = HD = 128 -> 4 frags, kept live).
  v16bf qf[4];
  #pragma unroll
  for (int ks = 0; ks < 4; ++ks)
    qf[ks] = load_a_frag(Qb + (size_t)(q0 + c) * EMB + hoff + ks * 32, lane);

  float m[8], l[8];
  #pragma unroll
  for (int j = 0; j < 8; ++j) { m[j] = -3.0e38f; l[j] = 0.f; }
  v8f o[8] = {};

#if HAVE_TDM
  const u16* kbase = Kb + hoff;
  if (w == 0)
    tdm_load_tile((unsigned)(uintptr_t)(void*)&Kt[0][0][0], kbase, HD, 32, EMB);
#endif

  for (int it = 0; it < SEQ / 32; ++it) {
    const int kv  = it * 32;
    const int cur = it & 1;
    __syncthreads();  // all waves done with previous tile buffers
#if HAVE_TDM
    if (w == 0 && kv + 32 < SEQ)
      tdm_load_tile((unsigned)(uintptr_t)(void*)&Kt[cur ^ 1][0][0],
                    kbase + (size_t)(kv + 32) * EMB, HD, 32, EMB);
#endif
    // Cooperative V load with transpose (and K when no TDM).
    #pragma unroll
    for (int i = 0; i < 16; ++i) {
      int e  = tid + 256 * i;    // 0..4095 over the 32x128 tile
      int kk = e >> 7;
      int dd = e & 127;
      Vt[dd][kk] = Vb[(size_t)(kv + kk) * EMB + hoff + dd];
#if !HAVE_TDM
      Kt[cur][kk][dd] = Kb[(size_t)(kv + kk) * EMB + hoff + dd];
#endif
    }
#if HAVE_TDM
    if (w == 0) { if (kv + 32 < SEQ) tdm_wait(1); else tdm_wait(0); }
#endif
    if (kv + 32 < SEQ) {  // prefetch next V tile (global_prefetch_b8)
      int kk = tid >> 3, dd = (tid & 7) << 4;
      __builtin_prefetch(Vb + (size_t)(kv + 32 + kk) * EMB + hoff + dd, 0, 1);
    }
    __syncthreads();

    // S = Q K^T : two 16-key column tiles. B[d][k'] = K[kv+k'][d]:
    // column k' = lane&15 -> Kt row, d contiguous.
    v8f s0 = {}, s1 = {};
    #pragma unroll
    for (int ks = 0; ks < 4; ++ks) {
      v16bf kb0 = load_b_frag(&Kt[cur][ 0 + c][ks * 32 + bsel]);
      v16bf kb1 = load_b_frag(&Kt[cur][16 + c][ks * 32 + bsel]);
      s0 = wmma_bf16(qf[ks], kb0, s0);
      s1 = wmma_bf16(qf[ks], kb1, s1);
    }

    // Online softmax: row j+half8 lives in this 16-lane half; DPP butterflies.
    u16* P = &Pp[w][0];
    #pragma unroll
    for (int j = 0; j < 8; ++j) {
      const int qr = q0 + j + half8;
      float a0 = s0[j] * SCALE + mask[(size_t)qr * SEQ + kv + c];
      float a1 = s1[j] * SCALE + mask[(size_t)qr * SEQ + kv + 16 + c];
      float mx   = red16_max(fmaxf(a0, a1));
      float mnew = fmaxf(m[j], mx);
      float corr = __expf(m[j] - mnew);
      float p0 = __expf(a0 - mnew);
      float p1 = __expf(a1 - mnew);
      float rs = red16_sum(p0 + p1);
      l[j] = l[j] * corr + rs;
      m[j] = mnew;
      #pragma unroll
      for (int t = 0; t < 8; ++t) o[t][j] *= corr;
      P[(j + half8) * 32 + c]      = f2bf(p0);   // C-layout -> row-major LDS
      P[(j + half8) * 32 + 16 + c] = f2bf(p1);
    }

    // O += P V : A = P re-read from LDS in A-fragment layout,
    // B = V per 16-wide d tile; Vt[d][k] gives contiguous-K b128 reads.
    v16bf pf = load_a_frag(P + (size_t)c * 32, lane);
    #pragma unroll
    for (int t = 0; t < 8; ++t) {
      v16bf vf = load_b_frag(&Vt[t * 16 + c][bsel]);
      o[t] = wmma_bf16(pf, vf, o[t]);
    }
  }

  // Normalize and write context slice (bf16) for the final projection.
  #pragma unroll
  for (int t = 0; t < 8; ++t)
    #pragma unroll
    for (int j = 0; j < 8; ++j) {
      float val = o[t][j] / l[j];
      Ctx[(size_t)(q0 + j + half8) * EMB + hoff + t * 16 + c] = f2bf(val);
    }
}

// ------------------------------------------------------------------- launcher
extern "C" void kernel_launch(void* const* d_in, const int* in_sizes, int n_in,
                              void* d_out, int out_size, void* d_ws, size_t ws_size,
                              hipStream_t stream) {
  (void)in_sizes; (void)n_in; (void)out_size; (void)ws_size;
  const float* X    = (const float*)d_in[0];
  const float* mask = (const float*)d_in[1];
  const float* Wq   = (const float*)d_in[2];
  const float* bq   = (const float*)d_in[3];
  const float* Wk   = (const float*)d_in[4];
  const float* bk   = (const float*)d_in[5];
  const float* Wv   = (const float*)d_in[6];
  const float* bv   = (const float*)d_in[7];
  const float* Wo   = (const float*)d_in[8];
  const float* bo   = (const float*)d_in[9];

  const size_t T = (size_t)SEQ * EMB;  // 4M elems (8 MB as bf16)
  u16* Xb  = (u16*)d_ws;               // workspace: 9 * 8 MB = 72 MB
  u16* Wqb = Xb  + T;
  u16* Wkb = Wqb + T;
  u16* Wvb = Wkb + T;
  u16* Wob = Wvb + T;
  u16* Qb  = Wob + T;
  u16* Kbf = Qb  + T;
  u16* Vbf = Kbf + T;
  u16* Cx  = Vbf + T;

  const int n = (int)T;
  dim3 cgrid(n / (256 * 4));
  cvt_f32_bf16<<<cgrid, 256, 0, stream>>>(X,  Xb,  n);
  cvt_f32_bf16<<<cgrid, 256, 0, stream>>>(Wq, Wqb, n);
  cvt_f32_bf16<<<cgrid, 256, 0, stream>>>(Wk, Wkb, n);
  cvt_f32_bf16<<<cgrid, 256, 0, stream>>>(Wv, Wvb, n);
  cvt_f32_bf16<<<cgrid, 256, 0, stream>>>(Wo, Wob, n);

  dim3 ggrid(EMB / 128, SEQ / 128);
  gemm_xt_wt<true ><<<ggrid, 256, 0, stream>>>(Xb, Wqb, bq, Qb,  SEQ, EMB, EMB);
  gemm_xt_wt<true ><<<ggrid, 256, 0, stream>>>(Xb, Wkb, bk, Kbf, SEQ, EMB, EMB);
  gemm_xt_wt<true ><<<ggrid, 256, 0, stream>>>(Xb, Wvb, bv, Vbf, SEQ, EMB, EMB);

  dim3 agrid(SEQ / 128, NH);
  attn_flash<<<agrid, 256, 0, stream>>>(Qb, Kbf, Vbf, mask, Cx);

  gemm_xt_wt<false><<<ggrid, 256, 0, stream>>>(Cx, Wob, bo, d_out, SEQ, EMB, EMB);
}
